// FuzzyAttention_51144470560870
// MI455X (gfx1250) — matrix operand order repaired and checked by reading, not compiled, not run on hardware
//
#include <hip/hip_runtime.h>

typedef __attribute__((ext_vector_type(16))) _Float16 v16h;
typedef __attribute__((ext_vector_type(8)))  _Float16 v8h;
typedef __attribute__((ext_vector_type(8)))  float    v8f;

#define WMMA_F16(A, B, C) \
  __builtin_amdgcn_wmma_f32_16x16x32_f16(false, (A), false, (B), (short)0, (C), false, false)

#define CAT16(LO, HI) \
  __builtin_shufflevector((LO), (HI), 0, 1, 2, 3, 4, 5, 6, 7, 8, 9, 10, 11, 12, 13, 14, 15)

constexpr int   Bb      = 2;
constexpr int   Ls      = 2048;
constexpr int   Hh      = 16;
constexpr int   Ee      = 64;
constexpr float SCALE   = 0.125f;     // 1/sqrt(64)
constexpr float DSCALE  = 0.0625f;    // DELTA * SCALE = 0.5 * 0.125
constexpr float NEGINF  = -1e30f;

__global__ __launch_bounds__(256)
void fuzzy_attn_kernel(const float* __restrict__ Q,
                       const float* __restrict__ K,
                       const float* __restrict__ V,
                       float* __restrict__ Out)
{
    // K tile:  [key][e]       stride 72 f16 = 144B (16B-aligned rows, bank stride 36 -> conflict-free)
    // V tile:  [e][key] (T)   stride 40 f16 =  80B (16B-aligned rows, bank stride 20 -> conflict-free)
    // P tiles: per-wave 16x32 stride 40 f16 =  80B
    __shared__ _Float16 k_lds[32][72];
    __shared__ _Float16 vt_lds[64][40];
    __shared__ _Float16 p_lds[8][16][40];

    const int tid    = threadIdx.x;
    const int wave   = tid >> 5;
    const int lane   = tid & 31;
    const int lane16 = lane & 15;
    const int half   = lane >> 4;

    const int bh      = blockIdx.y;
    const int b       = bh / Hh;
    const int head    = bh % Hh;
    const int q_block = blockIdx.x * 128;
    const int q_tile  = q_block + wave * 16;

    const long rowstride = (long)Hh * Ee;                  // 1024 floats between seq positions
    const float* Qbase = Q   + ((long)b * Ls * Hh + head) * Ee;
    const float* Kbase = K   + ((long)b * Ls * Hh + head) * Ee;
    const float* Vbase = V   + ((long)b * Ls * Hh + head) * Ee;
    float*       Obase = Out + ((long)b * Ls * Hh + head) * Ee;

    // cooperative-load thread mappings (block-wide, wave-independent)
    const int ck  = tid >> 3;              // K stage: key row 0..31
    const int ce0 = (tid & 7) * 8;         // K stage: e offset 0..56
    const int vkp = tid & 15;              // V stage: key pair -> keys 2vkp, 2vkp+1
    const int ve0 = (tid >> 4) * 4;        // V stage: e offset 0..60

    // ---- load Q A-fragments (16x32 f16, two K-chunks covering E=64) ----
    // A layout: lane%16 = row M; elements [8g+t] = K-dim index 32c + 16*half + 8g + t
    v16h aq0, aq1;
    {
        const float* qrow = Qbase + (long)(q_tile + lane16) * rowstride;
        #pragma unroll
        for (int c = 0; c < 2; ++c) {
            v16h a;
            #pragma unroll
            for (int g = 0; g < 2; ++g) {
                const float* p = qrow + 32 * c + 8 * g + 16 * half;
                #pragma unroll
                for (int t = 0; t < 8; ++t) a[8 * g + t] = (_Float16)p[t];
            }
            if (c == 0) aq0 = a; else aq1 = a;
        }
    }

    // online-softmax state (rows M = j + 8*half, replicated across the 16 lanes of a half)
    float m_run[8], l_run[8];
    v8f o0 = {}, o1 = {}, o2 = {}, o3 = {};
    #pragma unroll
    for (int j = 0; j < 8; ++j) { m_run[j] = NEGINF; l_run[j] = 0.0f; }

    const int n_tiles = (q_block >> 5) + 4;   // block-uniform: keys up to q_block+127

    for (int kt = 0; kt < n_tiles; ++kt) {
        const int k_base = kt * 32;

        __syncthreads();   // previous tile's LDS fully consumed

        // ---- cooperative stage: K tile -> k_lds (f32->f16 once per block) ----
        {
            const float* src = Kbase + (long)(k_base + ck) * rowstride + ce0;
            __builtin_prefetch(src + 32 * rowstride, 0, 1);   // next tile
            v8h h;
            #pragma unroll
            for (int t = 0; t < 8; ++t) h[t] = (_Float16)src[t];
            *(v8h*)&k_lds[ck][ce0] = h;
        }
        // ---- cooperative stage: V tile transposed -> vt_lds ----
        {
            const float* s0 = Vbase + (long)(k_base + 2 * vkp) * rowstride + ve0;
            const float* s1 = s0 + rowstride;
            __builtin_prefetch(s0 + 32 * rowstride, 0, 1);
            #pragma unroll
            for (int j = 0; j < 4; ++j) {
                union { _Float16 h[2]; unsigned u; } pk;
                pk.h[0] = (_Float16)s0[j];
                pk.h[1] = (_Float16)s1[j];
                *(unsigned*)&vt_lds[ve0 + j][2 * vkp] = pk.u;   // conflict-free packed store
            }
        }

        __syncthreads();   // tiles visible to all waves

        // ---- K B-fragments from LDS (lane%16 = key col; element t = e 16*half + t) ----
        const _Float16* r0 = &k_lds[lane16][16 * half];        // keys 0..15
        const _Float16* r1 = &k_lds[16 + lane16][16 * half];   // keys 16..31
        v16h bk00 = CAT16(*(const v8h*)(r0),      *(const v8h*)(r0 + 8));
        v16h bk01 = CAT16(*(const v8h*)(r0 + 32), *(const v8h*)(r0 + 40));
        v16h bk10 = CAT16(*(const v8h*)(r1),      *(const v8h*)(r1 + 8));
        v16h bk11 = CAT16(*(const v8h*)(r1 + 32), *(const v8h*)(r1 + 40));

        // ---- V B-fragments from LDS (lane%16 = e col; element t = key 16*half + t) ----
        const _Float16* rv0 = &vt_lds[lane16][16 * half];
        const _Float16* rv1 = &vt_lds[16 + lane16][16 * half];
        const _Float16* rv2 = &vt_lds[32 + lane16][16 * half];
        const _Float16* rv3 = &vt_lds[48 + lane16][16 * half];
        v16h bv0 = CAT16(*(const v8h*)(rv0), *(const v8h*)(rv0 + 8));
        v16h bv1 = CAT16(*(const v8h*)(rv1), *(const v8h*)(rv1 + 8));
        v16h bv2 = CAT16(*(const v8h*)(rv2), *(const v8h*)(rv2 + 8));
        v16h bv3 = CAT16(*(const v8h*)(rv3), *(const v8h*)(rv3 + 8));

        // ---- scores S = Q * K^T  (16x32, f32 accum) ----
        v8f s0 = {}, s1 = {};
        s0 = WMMA_F16(aq0, bk00, s0);
        s0 = WMMA_F16(aq1, bk01, s0);
        s1 = WMMA_F16(aq0, bk10, s1);
        s1 = WMMA_F16(aq1, bk11, s1);

        // ---- fuzzy transform + causal mask + online softmax ----
        // scale*(x + DELTA*sigmoid(x)) = SCALE*x + DSCALE*sigmoid(x); sigmoid via v_rcp_f32
        float v0[8], v1[8];
        #pragma unroll
        for (int j = 0; j < 8; ++j) {
            const int qrow = q_tile + j + 8 * half;     // C layout: VGPR j -> row j + 8*(lane/16)
            float x0   = s0[j];
            float sig0 = __builtin_amdgcn_rcpf(1.0f + __expf(-x0));
            float t0   = fmaf(DSCALE, sig0, SCALE * x0);
            v0[j] = (k_base + lane16 <= qrow) ? t0 : NEGINF;
            float x1   = s1[j];
            float sig1 = __builtin_amdgcn_rcpf(1.0f + __expf(-x1));
            float t1   = fmaf(DSCALE, sig1, SCALE * x1);
            v1[j] = (k_base + 16 + lane16 <= qrow) ? t1 : NEGINF;
        }

        float alpha[8];
        #pragma unroll
        for (int j = 0; j < 8; ++j) {
            float t = fmaxf(v0[j], v1[j]);
            t = fmaxf(t, __shfl_xor(t, 1, 32));
            t = fmaxf(t, __shfl_xor(t, 2, 32));
            t = fmaxf(t, __shfl_xor(t, 4, 32));
            t = fmaxf(t, __shfl_xor(t, 8, 32));   // masks <16 keep halves independent
            float m_new = fmaxf(m_run[j], t);
            alpha[j] = __expf(m_run[j] - m_new);
            m_run[j] = m_new;
        }

        float p0[8], p1[8];
        #pragma unroll
        for (int j = 0; j < 8; ++j) {
            p0[j] = __expf(v0[j] - m_run[j]);
            p1[j] = __expf(v1[j] - m_run[j]);
            float t = p0[j] + p1[j];
            t += __shfl_xor(t, 1, 32);
            t += __shfl_xor(t, 2, 32);
            t += __shfl_xor(t, 4, 32);
            t += __shfl_xor(t, 8, 32);
            l_run[j] = l_run[j] * alpha[j] + t;
        }

        #pragma unroll
        for (int j = 0; j < 8; ++j) {
            o0[j] *= alpha[j]; o1[j] *= alpha[j];
            o2[j] *= alpha[j]; o3[j] *= alpha[j];
        }

        // ---- transpose P (C layout -> A layout) through LDS ----
        #pragma unroll
        for (int j = 0; j < 8; ++j) {
            const int r = j + 8 * half;
            p_lds[wave][r][lane16]      = (_Float16)p0[j];
            p_lds[wave][r][16 + lane16] = (_Float16)p1[j];
        }
        __syncthreads();   // P visible (also: all waves done with k_lds/vt_lds reads)
        const _Float16* rp = &p_lds[wave][lane16][0];
        v16h ap = CAT16(*(const v8h*)(rp + 16 * half), *(const v8h*)(rp + 16 * half + 8));

        // ---- O += P * V ----
        o0 = WMMA_F16(ap, bv0, o0);
        o1 = WMMA_F16(ap, bv1, o1);
        o2 = WMMA_F16(ap, bv2, o2);
        o3 = WMMA_F16(ap, bv3, o3);
    }

    // ---- normalize and write O: [B, L, H, E] ----
    #pragma unroll
    for (int j = 0; j < 8; ++j) {
        const float inv = __builtin_amdgcn_rcpf(l_run[j]);
        const long  row = q_tile + j + 8 * half;
        float* orow = Obase + row * rowstride + lane16;
        orow[0]  = o0[j] * inv;
        orow[16] = o1[j] * inv;
        orow[32] = o2[j] * inv;
        orow[48] = o3[j] * inv;
    }
}

extern "C" void kernel_launch(void* const* d_in, const int* in_sizes, int n_in,
                              void* d_out, int out_size, void* d_ws, size_t ws_size,
                              hipStream_t stream) {
    const float* Q = (const float*)d_in[0];
    const float* K = (const float*)d_in[1];
    const float* V = (const float*)d_in[2];
    // d_in[3] is the triangular causal mask; it is applied analytically in-kernel.
    float* O = (float*)d_out;

    dim3 grid(Ls / 128, Bb * Hh);   // 16 x 32 blocks
    fuzzy_attn_kernel<<<grid, 256, 0, stream>>>(Q, K, V, O);
}
